// MultiHeadAttention_83640193122524
// MI455X (gfx1250) — matrix operand order repaired and checked by reading, not compiled, not run on hardware
//
#include <hip/hip_runtime.h>

typedef __attribute__((ext_vector_type(16))) _Float16 v16h;
typedef __attribute__((ext_vector_type(8)))  float    v8f;
typedef __attribute__((ext_vector_type(4)))  unsigned int v4u;
typedef __attribute__((ext_vector_type(4)))  float    v4f;

#define B_  8
#define T_  2048
#define C_  256
#define H_  8
#define D_  32
#define HD_ 256   // H_*D_

union Frag {
  v16h v;
  v4u  q[2];
  _Float16 h[16];
};

__device__ __forceinline__ v8f wmma_f16(const Frag& a, const Frag& b, v8f c) {
  return __builtin_amdgcn_wmma_f32_16x16x32_f16(false, a.v, false, b.v,
                                                (short)0, c, false, false);
}

// ---------------------------------------------------------------------------
// Kernel 1: QKV projection.  grid = (B*T/64, 3), block = 256 (8 waves).
// z selects {Q,K,V}.  Per block: 64 t-rows, wave w = head h, 32 d-cols.
// Q,K stored [b,h,t,d] f16; V stored transposed [b,h,d,t] f16.
// ---------------------------------------------------------------------------
__global__ __launch_bounds__(256) void qkv_kernel(
    const float* __restrict__ x, const float* __restrict__ wq,
    const float* __restrict__ wk, const float* __restrict__ wv,
    _Float16* __restrict__ qbuf, _Float16* __restrict__ kbuf,
    _Float16* __restrict__ vtbuf)
{
  constexpr int XS = 264;                 // padded row stride (halves)
  __shared__ _Float16 xs[64 * XS];
  const int tile = blockIdx.x;
  const int z    = blockIdx.y;
  const int b    = tile >> 5;             // / (T/64)
  const int t0   = (tile & 31) << 6;      // * 64
  const float* w = (z == 0) ? wq : (z == 1) ? wk : wv;

  const int tid = threadIdx.x;
  {
    // cooperative load x[b][t0:t0+64][:] -> f16 LDS tile
    const int row = tid >> 2;
    const int cb  = (tid & 3) << 6;
    const v4f* xg = (const v4f*)(x + ((size_t)b * T_ + t0 + row) * C_ + cb);
    _Float16* xr = xs + row * XS + cb;
#pragma unroll
    for (int j = 0; j < 16; ++j) {
      v4f f = xg[j];
      xr[4*j+0] = (_Float16)f.x; xr[4*j+1] = (_Float16)f.y;
      xr[4*j+2] = (_Float16)f.z; xr[4*j+3] = (_Float16)f.w;
    }
  }
  __syncthreads();

  const int h    = tid >> 5;              // wave id == head
  const int lane = tid & 31;
  const int li = lane & 15, hi = lane >> 4;

  v8f acc[4][2];
#pragma unroll
  for (int mt = 0; mt < 4; ++mt)
#pragma unroll
    for (int nt = 0; nt < 2; ++nt) acc[mt][nt] = (v8f){};

#pragma unroll
  for (int ks = 0; ks < 8; ++ks) {
    Frag a[4];
#pragma unroll
    for (int mt = 0; mt < 4; ++mt) {
      const _Float16* base = xs + (mt*16 + li) * XS + ks*32;
      a[mt].q[0] = *(const v4u*)(base + 16*hi);
      a[mt].q[1] = *(const v4u*)(base + 16*hi + 8);
    }
    Frag bf[2];
#pragma unroll
    for (int nt = 0; nt < 2; ++nt) {
      const int d = nt*16 + li;
      const float* wp = w + ((size_t)h * C_ + ks*32 + 16*hi) * D_ + d;
#pragma unroll
      for (int e = 0; e < 16; ++e) bf[nt].h[e] = (_Float16)wp[(size_t)e * D_];
    }
#pragma unroll
    for (int mt = 0; mt < 4; ++mt)
#pragma unroll
      for (int nt = 0; nt < 2; ++nt)
        acc[mt][nt] = wmma_f16(a[mt], bf[nt], acc[mt][nt]);
  }

  const size_t bh = (size_t)b * H_ + h;
  if (z < 2) {
    _Float16* outp = (z == 0 ? qbuf : kbuf) + bh * T_ * D_;
#pragma unroll
    for (int mt = 0; mt < 4; ++mt)
#pragma unroll
      for (int nt = 0; nt < 2; ++nt) {
        const int d = nt*16 + li;
#pragma unroll
        for (int r = 0; r < 8; ++r) {
          const int t = t0 + mt*16 + 8*hi + r;
          outp[(size_t)t * D_ + d] = (_Float16)acc[mt][nt][r];
        }
      }
  } else {
    // transposed store: Vt[d][t]; 8 consecutive t per lane -> one b128 store
#pragma unroll
    for (int mt = 0; mt < 4; ++mt)
#pragma unroll
      for (int nt = 0; nt < 2; ++nt) {
        const int d = nt*16 + li;
        Frag pk;
#pragma unroll
        for (int r = 0; r < 8; ++r) pk.h[r] = (_Float16)acc[mt][nt][r];
        _Float16* vp = vtbuf + (bh * D_ + d) * T_ + t0 + mt*16 + 8*hi;
        *(v4u*)vp = pk.q[0];
      }
  }
}

// ---------------------------------------------------------------------------
// Kernel 2: flash attention.  grid = B*H*(T/128), block = 256 (8 waves).
// wave owns 16 q-rows; streams s in chunks of 32 with online softmax.
// ---------------------------------------------------------------------------
__global__ __launch_bounds__(256) void attn_kernel(
    const _Float16* __restrict__ qbuf, const _Float16* __restrict__ kbuf,
    const _Float16* __restrict__ vtbuf, _Float16* __restrict__ abuf)
{
  constexpr int PS = 40;                  // padded P-tile row stride (halves)
  __shared__ _Float16 pb[8][16 * PS];
  const int blk  = blockIdx.x;
  const int bh   = blk >> 4;              // / (T/128)
  const int qblk = blk & 15;
  const int b = bh >> 3, h = bh & 7;

  const int tid  = threadIdx.x;
  const int wv   = tid >> 5;
  const int lane = tid & 31;
  const int li = lane & 15, hi = lane >> 4;
  const int q0 = qblk * 128 + wv * 16;
  _Float16* pw = pb[wv];

  Frag aq;
  {
    const _Float16* qp = qbuf + ((size_t)bh * T_ + q0 + li) * D_;
    aq.q[0] = *(const v4u*)(qp + 16*hi);
    aq.q[1] = *(const v4u*)(qp + 16*hi + 8);
  }

  const _Float16* kbase = kbuf  + (size_t)bh * T_ * D_;
  const _Float16* vbase = vtbuf + (size_t)bh * D_ * T_;

  v8f o0 = (v8f){}, o1 = (v8f){};
  float m[8], l[8];
#pragma unroll
  for (int r = 0; r < 8; ++r) { m[r] = -1e30f; l[r] = 0.0f; }

  for (int s0 = 0; s0 < T_; s0 += 32) {
    Frag bk0, bk1;
    const _Float16* kp0 = kbase + (size_t)(s0 + li) * D_ + 16*hi;
    const _Float16* kp1 = kbase + (size_t)(s0 + 16 + li) * D_ + 16*hi;
    bk0.q[0] = ((const v4u*)kp0)[0]; bk0.q[1] = ((const v4u*)kp0)[1];
    bk1.q[0] = ((const v4u*)kp1)[0]; bk1.q[1] = ((const v4u*)kp1)[1];
    if (s0 + 32 < T_)
      __builtin_prefetch(kbase + (size_t)(s0 + 32 + li) * D_, 0, 0);

    v8f zz = (v8f){};
    v8f sA = wmma_f16(aq, bk0, zz);       // S[:, s0    .. s0+16)
    v8f sB = wmma_f16(aq, bk1, zz);       // S[:, s0+16 .. s0+32)

    float p0[8], p1[8];
#pragma unroll
    for (int r = 0; r < 8; ++r) {
      float v0 = sA[r] * 0.0625f;         // 1/sqrt(C), C=256
      float v1 = sB[r] * 0.0625f;
      float mx = fmaxf(v0, v1);
      mx = fmaxf(mx, __shfl_xor(mx, 1, 32));
      mx = fmaxf(mx, __shfl_xor(mx, 2, 32));
      mx = fmaxf(mx, __shfl_xor(mx, 4, 32));
      mx = fmaxf(mx, __shfl_xor(mx, 8, 32));
      float mn = fmaxf(m[r], mx);
      float al = __expf(m[r] - mn);
      p0[r] = __expf(v0 - mn);
      p1[r] = __expf(v1 - mn);
      float sum = p0[r] + p1[r];
      sum += __shfl_xor(sum, 1, 32);
      sum += __shfl_xor(sum, 2, 32);
      sum += __shfl_xor(sum, 4, 32);
      sum += __shfl_xor(sum, 8, 32);
      l[r] = l[r] * al + sum;
      o0[r] *= al;
      o1[r] *= al;
      m[r] = mn;
    }

    // lane-transpose P through LDS: C-layout -> A-layout
#pragma unroll
    for (int r = 0; r < 8; ++r) {
      const int t = 8*hi + r;
      pw[t * PS + li]      = (_Float16)p0[r];
      pw[t * PS + 16 + li] = (_Float16)p1[r];
    }
    asm volatile("s_wait_dscnt 0" ::: "memory");

    Frag ap;
    ap.q[0] = *(const v4u*)(pw + li * PS + 16*hi);
    ap.q[1] = *(const v4u*)(pw + li * PS + 16*hi + 8);

    Frag bv0, bv1;
    const _Float16* vp0 = vbase + (size_t)li * T_ + s0 + 16*hi;
    const _Float16* vp1 = vbase + (size_t)(16 + li) * T_ + s0 + 16*hi;
    bv0.q[0] = ((const v4u*)vp0)[0]; bv0.q[1] = ((const v4u*)vp0)[1];
    bv1.q[0] = ((const v4u*)vp1)[0]; bv1.q[1] = ((const v4u*)vp1)[1];

    o0 = wmma_f16(ap, bv0, o0);
    o1 = wmma_f16(ap, bv1, o1);
  }

  // normalize + store to abuf[b][t][h*32+d] (f16)
#pragma unroll
  for (int r = 0; r < 8; ++r) {
    const float inv = 1.0f / l[r];
    const int t = q0 + 8*hi + r;
    _Float16* op = abuf + ((size_t)b * T_ + t) * HD_;
    op[h*32 + li]      = (_Float16)(o0[r] * inv);
    op[h*32 + 16 + li] = (_Float16)(o1[r] * inv);
  }
}

// ---------------------------------------------------------------------------
// Kernel 3: output projection out = attn @ wproj + bproj (fp32 out).
// grid = B*T/64, block = 256 (8 waves); wave w owns cols [32w, 32w+32).
// ---------------------------------------------------------------------------
__global__ __launch_bounds__(256) void proj_kernel(
    const _Float16* __restrict__ abuf, const float* __restrict__ wproj,
    const float* __restrict__ bproj, float* __restrict__ out)
{
  constexpr int XS = 264;
  __shared__ _Float16 as_[64 * XS];
  const int tile = blockIdx.x;
  const int b  = tile >> 5;
  const int t0 = (tile & 31) << 6;
  const int tid = threadIdx.x;
  {
    const int row = tid >> 2;
    const int cb  = (tid & 3) << 6;
    const v4u* ag = (const v4u*)(abuf + ((size_t)b * T_ + t0 + row) * HD_ + cb);
    v4u* ar = (v4u*)(as_ + row * XS + cb);
#pragma unroll
    for (int j = 0; j < 8; ++j) ar[j] = ag[j];
  }
  __syncthreads();

  const int wv   = tid >> 5;
  const int lane = tid & 31;
  const int li = lane & 15, hi = lane >> 4;
  const int c0 = wv * 32;

  v8f acc[4][2];
#pragma unroll
  for (int mt = 0; mt < 4; ++mt)
#pragma unroll
    for (int nt = 0; nt < 2; ++nt) acc[mt][nt] = (v8f){};

#pragma unroll
  for (int ks = 0; ks < 8; ++ks) {
    Frag a[4];
#pragma unroll
    for (int mt = 0; mt < 4; ++mt) {
      const _Float16* base = as_ + (mt*16 + li) * XS + ks*32;
      a[mt].q[0] = *(const v4u*)(base + 16*hi);
      a[mt].q[1] = *(const v4u*)(base + 16*hi + 8);
    }
    Frag bf[2];
#pragma unroll
    for (int nt = 0; nt < 2; ++nt) {
      const int col = c0 + nt*16 + li;
      const float* wp = wproj + (size_t)(ks*32 + 16*hi) * C_ + col;
#pragma unroll
      for (int e = 0; e < 16; ++e) bf[nt].h[e] = (_Float16)wp[(size_t)e * C_];
    }
#pragma unroll
    for (int mt = 0; mt < 4; ++mt)
#pragma unroll
      for (int nt = 0; nt < 2; ++nt)
        acc[mt][nt] = wmma_f16(a[mt], bf[nt], acc[mt][nt]);
  }

  const float b0 = bproj[c0 + li];
  const float b1 = bproj[c0 + 16 + li];
#pragma unroll
  for (int mt = 0; mt < 4; ++mt)
#pragma unroll
    for (int r = 0; r < 8; ++r) {
      const int t = t0 + mt*16 + 8*hi + r;
      float* op = out + ((size_t)b * T_ + t) * C_;
      op[c0 + li]      = acc[mt][0][r] + b0;
      op[c0 + 16 + li] = acc[mt][1][r] + b1;
    }
}

// ---------------------------------------------------------------------------
extern "C" void kernel_launch(void* const* d_in, const int* in_sizes, int n_in,
                              void* d_out, int out_size, void* d_ws, size_t ws_size,
                              hipStream_t stream) {
  (void)in_sizes; (void)n_in; (void)out_size; (void)ws_size;
  const float* x     = (const float*)d_in[0];
  const float* wq    = (const float*)d_in[1];
  const float* wk    = (const float*)d_in[2];
  const float* wv    = (const float*)d_in[3];
  const float* wproj = (const float*)d_in[4];
  const float* bproj = (const float*)d_in[5];
  float* out = (float*)d_out;

  const size_t n = (size_t)B_ * H_ * T_ * D_;   // 4,194,304 elements
  _Float16* qbuf  = (_Float16*)d_ws;            // 8 MB
  _Float16* kbuf  = qbuf + n;                   // 8 MB
  _Float16* vtbuf = kbuf + n;                   // 8 MB (transposed [b,h,d,t])
  _Float16* abuf  = vtbuf + n;                  // 8 MB ([b,t,h*d])

  qkv_kernel<<<dim3(B_ * T_ / 64, 3), 256, 0, stream>>>(
      x, wq, wk, wv, qbuf, kbuf, vtbuf);
  attn_kernel<<<dim3(B_ * H_ * (T_ / 128)), 256, 0, stream>>>(
      qbuf, kbuf, vtbuf, abuf);
  proj_kernel<<<dim3(B_ * T_ / 64), 256, 0, stream>>>(
      abuf, wproj, bproj, out);
}
